// MultiHeadAttention_67104569032757
// MI455X (gfx1250) — matrix operand order repaired and checked by reading, not compiled, not run on hardware
//
#include <hip/hip_runtime.h>

typedef __attribute__((ext_vector_type(16))) __bf16 v16bf;
typedef __attribute__((ext_vector_type(8)))  float  v8f;
typedef unsigned short u16;
typedef unsigned int   u32;

// Problem dims (fixed by reference)
constexpr int GB  = 2;
constexpr int GS  = 2048;
constexpr int GDIM = 2048;
constexpr int GH  = 16;
constexpr int GHD = 128;
constexpr int GM  = GB * GS;   // 4096 rows for all GEMMs
constexpr int GK  = GDIM;      // 2048
constexpr int GN  = GDIM;      // 2048

// ---------------- CDNA5 feature detection (device pass only) -------------
#if defined(__AMDGCN__) && __has_builtin(__builtin_amdgcn_global_load_async_to_lds_b128) && __has_builtin(__builtin_amdgcn_s_wait_asynccnt)
#define USE_ASYNC 1
#else
#define USE_ASYNC 0
#endif

#if defined(__AMDGCN__) && __has_builtin(__builtin_amdgcn_tensor_load_to_lds) && __has_builtin(__builtin_amdgcn_s_wait_tensorcnt)
#define USE_TDM 1
#else
#define USE_TDM 0
#endif

#if USE_TDM || USE_ASYNC
typedef unsigned int uint32x4 __attribute__((ext_vector_type(4)));
typedef int          int32x4  __attribute__((ext_vector_type(4)));
typedef int          int32x8  __attribute__((ext_vector_type(8)));
#endif

// ---------- helpers ----------
__device__ __forceinline__ u16 f2bf(float f) {
  u32 u = __float_as_uint(f);
  u += 0x7FFFu + ((u >> 16) & 1u);          // round-to-nearest-even
  return (u16)(u >> 16);
}
__device__ __forceinline__ u32 pack2bf(float x, float y) {
  return ((u32)f2bf(y) << 16) | (u32)f2bf(x);
}
__device__ __forceinline__ void store_elem(float* C, size_t off, float v) { C[off] = v; }
__device__ __forceinline__ void store_elem(u16*   C, size_t off, float v) { C[off] = f2bf(v); }

template <typename T> struct ARegT;
template <> struct ARegT<float> { using T = float4; };
template <> struct ARegT<u16>   { using T = uint2;  };
template <typename T> struct IsBF16 { static constexpr bool value = false; };
template <> struct IsBF16<u16>      { static constexpr bool value = true;  };

__device__ __forceinline__ float4 load_a_raw(const float* p) { return *(const float4*)p; }
__device__ __forceinline__ uint2  load_a_raw(const u16*   p) { return *(const uint2*)p; }
__device__ __forceinline__ uint2  to_bf4(float4 f) { return make_uint2(pack2bf(f.x, f.y), pack2bf(f.z, f.w)); }
__device__ __forceinline__ uint2  to_bf4(uint2 d)  { return d; }

union F16x16 {            // one 16x(16-bit) WMMA operand fragment (8 VGPRs)
  v16bf v;
  uint4 q[2];
};

__device__ __forceinline__ v8f wmma_bf16(const v16bf& a, const v16bf& b, const v8f& c) {
  return __builtin_amdgcn_wmma_f32_16x16x32_bf16(false, a, false, b, (short)0, c,
                                                 false, false);
}

// ---------------- async / TDM staging primitives -------------------------
__device__ __forceinline__ void g2l_b128(u16* l, const u16* g) {
#if USE_ASYNC
  __builtin_amdgcn_global_load_async_to_lds_b128(
      (__attribute__((address_space(1))) int32x4*)g,
      (__attribute__((address_space(3))) int32x4*)l, 0, 0);
#else
  *(uint4*)l = *(const uint4*)g;
#endif
}

#if USE_TDM
__device__ __forceinline__ u32 lds_off(const void* p) {
  return (u32)(unsigned long long)(__attribute__((address_space(3))) const void*)p;
}
// 2D bf16 tile load via Tensor Data Mover. tile_w elements per row (must be
// 64 DWORDs wide here: pad_interval=5 inserts pad_amount=4 DWORDs -> LDS row
// stride = 128+8 halves, matching LQK below.
__device__ __forceinline__ void tdm_load_2d(u32 ldsoff, const u16* g,
                                            int tile_w, int tile_h, int stride_e) {
  unsigned long long ga = (unsigned long long)g;
  uint32x4 g0;
  g0[0] = 1u;                                            // count=1 (valid D#)
  g0[1] = ldsoff;                                        // lds_addr (bytes)
  g0[2] = (u32)ga;                                       // global_addr lo
  g0[3] = (u32)((ga >> 32) & 0x1FFFFFFull) | (2u << 30); // addr hi | type=2
  int32x8 g1;
  g1[0] = (int)((1u << 16) | (1u << 20) | (5u << 22) | (3u << 25)); // 2B, pad
  g1[1] = (int)(((u32)tile_w & 0xFFFFu) << 16);          // tensor_dim0 lo16
  g1[2] = (int)(((u32)tile_h & 0xFFFFu) << 16);          // tensor_dim1 lo16
  g1[3] = (int)(((u32)tile_w & 0xFFFFu) << 16);          // tile_dim0
  g1[4] = (int)((u32)tile_h & 0xFFFFu);                  // tile_dim1
  g1[5] = stride_e;                                      // tensor_dim0_stride
  g1[6] = 0; g1[7] = 0;
  int32x4 z4 = {0, 0, 0, 0};
#if __clang_major__ >= 23
  int32x8 z8 = {0, 0, 0, 0, 0, 0, 0, 0};
  __builtin_amdgcn_tensor_load_to_lds(g0, g1, z4, z4, z8, 0);
#else
  __builtin_amdgcn_tensor_load_to_lds(g0, g1, z4, z4, 0);
#endif
}
#endif

// =====================================================================
// GEMM: C[M,N] = A[M,K] @ W[K,N]; A f32|bf16, W f32 (converted), C bf16|f32.
// 128x128x32 block tiles, 256 thr (8 waves), wave tile 32x64, LDS double-
// buffered software pipeline; A staged via ASYNCcnt when already bf16.
// =====================================================================
constexpr int BM = 128, BN = 128, BK = 32;
constexpr int LDS_A = BK + 8;   // 40 halves (80B rows, 16B aligned)

template <typename TA, typename TC>
__global__ __launch_bounds__(256)
void gemm_wmma(const TA* __restrict__ A, const float* __restrict__ W,
               TC* __restrict__ C) {
  __shared__ u16 sA[2][BM * LDS_A];   // [m][k]
  __shared__ u16 sB[2][BN * LDS_A];   // [n][k] (transposed weight tile)

  const int tid  = threadIdx.x;
  const int wave = tid >> 5;
  const int lane = tid & 31;
  const int half = lane >> 4;
  const int l16  = lane & 15;
  const int m0 = blockIdx.y * BM;
  const int n0 = blockIdx.x * BN;
  const int wr = wave >> 1;        // 0..3 -> 32-row strip
  const int wc = wave & 1;         // 0..1 -> 64-col strip

  constexpr bool ASYNC_A = IsBF16<TA>::value && (USE_ASYNC != 0);

  v8f acc[2][4] = {};
  typename ARegT<TA>::T ra[4];
  float4 rw[4];

  auto issue_a = [&](int kb, int nbuf) {
    if constexpr (ASYNC_A) {
#if USE_ASYNC
#pragma unroll
      for (int i = 0; i < 2; ++i) {
        int c8 = tid + i * 256;               // 512 b128 chunks
        int r  = c8 >> 2;
        int cc = (c8 & 3) * 8;
        g2l_b128(&sA[nbuf][r * LDS_A + cc],
                 (const u16*)A + (size_t)(m0 + r) * GK + kb + cc);
      }
#endif
    } else {
#pragma unroll
      for (int i = 0; i < 4; ++i) {
        int q4 = tid + i * 256;
        int r  = q4 >> 3;
        int c  = (q4 & 7) * 4;
        ra[i] = load_a_raw(A + (size_t)(m0 + r) * GK + kb + c);
      }
    }
  };
  auto issue_w = [&](int kb) {
#pragma unroll
    for (int i = 0; i < 4; ++i) {
      int q4 = tid + i * 256;
      int k  = q4 >> 5;
      int n  = (q4 & 31) * 4;
      rw[i] = *(const float4*)(W + (size_t)(kb + k) * GN + n0 + n);
    }
  };
  auto commit = [&](int nbuf) {
    if constexpr (!ASYNC_A) {
#pragma unroll
      for (int i = 0; i < 4; ++i) {
        int q4 = tid + i * 256;
        int r  = q4 >> 3;
        int c  = (q4 & 7) * 4;
        *(uint2*)&sA[nbuf][r * LDS_A + c] = to_bf4(ra[i]);
      }
    }
#pragma unroll
    for (int i = 0; i < 4; ++i) {
      int q4 = tid + i * 256;
      int k  = q4 >> 5;
      int n  = (q4 & 31) * 4;
      float4 f = rw[i];
      sB[nbuf][(n + 0) * LDS_A + k] = f2bf(f.x);
      sB[nbuf][(n + 1) * LDS_A + k] = f2bf(f.y);
      sB[nbuf][(n + 2) * LDS_A + k] = f2bf(f.z);
      sB[nbuf][(n + 3) * LDS_A + k] = f2bf(f.w);
    }
  };
  auto stage_wait = [&]() {
#if USE_ASYNC
    __builtin_amdgcn_s_wait_asynccnt(0);
#endif
  };

  // ---- prologue ----
  issue_a(0, 0);
  issue_w(0);
  commit(0);
  stage_wait();
  __syncthreads();

  int buf = 0;
  for (int kb = 0; kb < GK; kb += BK) {
    const bool more = (kb + BK < GK);
    if (more) {
      issue_a(kb + BK, buf ^ 1);     // async path writes LDS directly
      issue_w(kb + BK);
      __builtin_prefetch(W + (size_t)(kb + 2 * BK + (tid & 31)) * GN + n0, 0, 3);
    }

    // ---- fragments + 8 WMMAs per wave on buffer `buf` ----
    F16x16 a[2];
#pragma unroll
    for (int mi = 0; mi < 2; ++mi) {
      int ar = (wr * 32 + mi * 16 + l16) * LDS_A;
      a[mi].q[0] = *(const uint4*)&sA[buf][ar + (half ? 8 : 0)];
      a[mi].q[1] = *(const uint4*)&sA[buf][ar + (half ? 24 : 16)];
    }
#pragma unroll
    for (int ni = 0; ni < 4; ++ni) {
      F16x16 b;
      int br = (wc * 64 + ni * 16 + l16) * LDS_A + (half ? 8 : 0);
      b.q[0] = *(const uint4*)&sB[buf][br];
      b.q[1] = *(const uint4*)&sB[buf][br + 16];
#pragma unroll
      for (int mi = 0; mi < 2; ++mi)
        acc[mi][ni] = wmma_bf16(a[mi].v, b.v, acc[mi][ni]);
    }

    if (more) commit(buf ^ 1);
    stage_wait();
    __syncthreads();
    buf ^= 1;
  }

  // ---- store: D layout = VGPR i -> row (i | i+8), lane%16 -> col ----
#pragma unroll
  for (int mi = 0; mi < 2; ++mi) {
#pragma unroll
    for (int ni = 0; ni < 4; ++ni) {
      int rbase = m0 + wr * 32 + mi * 16 + (half ? 8 : 0);
      int col   = n0 + wc * 64 + ni * 16 + l16;
#pragma unroll
      for (int i = 0; i < 8; ++i)
        store_elem(C, (size_t)(rbase + i) * GN + col, acc[mi][ni][i]);
    }
  }
}

// =====================================================================
// Flash attention (causal). Grid: (S/64, B*H). 128 threads = 4 waves,
// each wave owns 16 query rows. HD=128. Q/K tiles staged by the TDM
// (TENSORcnt) with LDS padding; K/V double-buffered against the WMMAs.
// =====================================================================
constexpr int QT = 64, KTILE = 64;
constexpr int LQK = GHD + 8;     // 136 halves (matches TDM pad descriptor)
constexpr int LVS = KTILE + 8;   // 72 halves

__global__ __launch_bounds__(128)
void flash_attn(const u16* __restrict__ xq, const u16* __restrict__ xk,
                const u16* __restrict__ xv, u16* __restrict__ out) {
  __shared__ u16 sQ[QT * LQK];        // [q][hd]
  __shared__ u16 sK[2][KTILE * LQK];  // [key][hd]
  __shared__ u16 sV[2][GHD * LVS];    // [hd][key] (transposed)
  __shared__ u16 sP[QT * LVS];        // [q][key] probs round-trip

  const int tid  = threadIdx.x;
  const int wave = tid >> 5;
  const int lane = tid & 31;
  const int half = lane >> 4;
  const int l16  = lane & 15;
  const int qt = blockIdx.x;
  const int bh = blockIdx.y;
  const int b  = bh >> 4;          // H = 16
  const int h  = bh & 15;

  const size_t headoff = ((size_t)b * GS) * GDIM + (size_t)h * GHD;
  const u16* gQ = xq + headoff + (size_t)qt * QT * GDIM;

  auto stage_qk = [&](u16* lds, const u16* g) {
#if USE_TDM
    if (wave == 0) tdm_load_2d(lds_off(lds), g, GHD, KTILE, GDIM);
#else
#pragma unroll
    for (int i = 0; i < 8; ++i) {
      int c  = tid + i * 128;
      int r  = c >> 4;
      int cc = (c & 15) * 8;
      g2l_b128(&lds[r * LQK + cc], g + (size_t)r * GDIM + cc);
    }
#endif
  };
  auto stage_wait = [&]() {
#if USE_TDM
    if (wave == 0) __builtin_amdgcn_s_wait_tensorcnt(0);
#elif USE_ASYNC
    __builtin_amdgcn_s_wait_asynccnt(0);
#endif
  };

  uint4 rv[8];
  auto issue_v = [&](int kt) {
    const u16* gV = xv + headoff + (size_t)kt * KTILE * GDIM;
#pragma unroll
    for (int i = 0; i < 8; ++i) {
      int c  = tid + i * 128;
      int r  = c >> 4;
      int cc = (c & 15) * 8;
      rv[i] = *(const uint4*)(gV + (size_t)r * GDIM + cc);
    }
  };
  auto commit_v = [&](int nbuf) {       // transpose scatter [key][hd]->[hd][key]
#pragma unroll
    for (int i = 0; i < 8; ++i) {
      int c  = tid + i * 128;
      int r  = c >> 4;
      int cc = (c & 15) * 8;
      const u16* p = (const u16*)&rv[i];
#pragma unroll
      for (int j = 0; j < 8; ++j) sV[nbuf][(cc + j) * LVS + r] = p[j];
    }
  };

  // ---- prologue: Q tile + first K/V tiles ----
  stage_qk(sQ, gQ);
  stage_qk(sK[0], xk + headoff);
  issue_v(0);
  commit_v(0);
  stage_wait();
  __syncthreads();

  // Q fragments, loaded once (A-layout, 16x32 per hd-chunk)
  F16x16 aQ[4];
  {
    int qr = (wave * 16 + l16) * LQK;
#pragma unroll
    for (int kc = 0; kc < 4; ++kc) {
      aQ[kc].q[0] = *(const uint4*)&sQ[qr + kc * 32 + (half ? 8 : 0)];
      aQ[kc].q[1] = *(const uint4*)&sQ[qr + kc * 32 + (half ? 24 : 16)];
    }
  }

  float mrow[8], lrow[8];
  v8f accO[8] = {};
#pragma unroll
  for (int i = 0; i < 8; ++i) { mrow[i] = -1e30f; lrow[i] = 0.f; }

  // 1/sqrt(128) * log2(e): softmax in exp2 domain
  const float sscale = 0.08838834764831845f * 1.44269504088896341f;

  const int nkt = qt + 1;            // causal: key tiles 0..qt
  int buf = 0;
  for (int kt = 0; kt < nkt; ++kt) {
    const bool more = (kt + 1 < nkt);
    if (more) {                       // stage next K via TDM, next V to regs
      stage_qk(sK[buf ^ 1], xk + headoff + (size_t)(kt + 1) * KTILE * GDIM);
      issue_v(kt + 1);
    }

    // ---- scores: 16x64 per wave = 4 col-tiles x 4 hd-chunks of WMMA ----
    v8f sc[4] = {};
#pragma unroll
    for (int ni = 0; ni < 4; ++ni) {
#pragma unroll
      for (int kc = 0; kc < 4; ++kc) {
        F16x16 bK;
        int br = (ni * 16 + l16) * LQK + kc * 32 + (half ? 8 : 0);
        bK.q[0] = *(const uint4*)&sK[buf][br];
        bK.q[1] = *(const uint4*)&sK[buf][br + 16];
        sc[ni] = wmma_bf16(aQ[kc].v, bK.v, sc[ni]);
      }
    }

    if (more) commit_v(buf ^ 1);     // frees rv before the VALU-heavy part

    // ---- scale + causal mask (diag tile only) ----
    const bool diag = (kt == qt);
    const int rbase = wave * 16 + (half ? 8 : 0);
#pragma unroll
    for (int ni = 0; ni < 4; ++ni) {
      int col = ni * 16 + l16;
#pragma unroll
      for (int i = 0; i < 8; ++i) {
        float s = sc[ni][i] * sscale;
        if (diag && col > rbase + i) s = -1e30f;
        sc[ni][i] = s;
      }
    }

    // ---- online softmax per row (rows live in 16-lane halves) ----
#pragma unroll
    for (int i = 0; i < 8; ++i) {
      float mx = mrow[i];
#pragma unroll
      for (int ni = 0; ni < 4; ++ni) mx = fmaxf(mx, sc[ni][i]);
#pragma unroll
      for (int d = 1; d < 16; d <<= 1) mx = fmaxf(mx, __shfl_xor(mx, d, 32));
      float alpha = exp2f(mrow[i] - mx);
      float rs = 0.f;
#pragma unroll
      for (int ni = 0; ni < 4; ++ni) {
        float p = exp2f(sc[ni][i] - mx);
        sc[ni][i] = p;
        rs += p;
      }
#pragma unroll
      for (int d = 1; d < 16; d <<= 1) rs += __shfl_xor(rs, d, 32);
      lrow[i] = lrow[i] * alpha + rs;
      mrow[i] = mx;
#pragma unroll
      for (int g = 0; g < 8; ++g) accO[g][i] *= alpha;
    }

    // ---- P -> LDS (bf16), wave-private region, C-layout -> A-layout ----
#pragma unroll
    for (int ni = 0; ni < 4; ++ni)
#pragma unroll
      for (int i = 0; i < 8; ++i)
        sP[(rbase + i) * LVS + ni * 16 + l16] = f2bf(sc[ni][i]);

    F16x16 aP[2];
    {
      int pr = (wave * 16 + l16) * LVS;
#pragma unroll
      for (int kk = 0; kk < 2; ++kk) {
        aP[kk].q[0] = *(const uint4*)&sP[pr + kk * 32 + (half ? 8 : 0)];
        aP[kk].q[1] = *(const uint4*)&sP[pr + kk * 32 + (half ? 24 : 16)];
      }
    }

    // ---- O += P @ V : 8 col groups x 2 key chunks ----
#pragma unroll
    for (int g = 0; g < 8; ++g) {
#pragma unroll
      for (int kk = 0; kk < 2; ++kk) {
        F16x16 bV;
        int br = (g * 16 + l16) * LVS + kk * 32 + (half ? 8 : 0);
        bV.q[0] = *(const uint4*)&sV[buf][br];
        bV.q[1] = *(const uint4*)&sV[buf][br + 16];
        accO[g] = wmma_bf16(aP[kk].v, bV.v, accO[g]);
      }
    }

    stage_wait();
    __syncthreads();
    buf ^= 1;
  }

  // ---- epilogue: normalize and store bf16 [b, s, h*HD + d] ----
  u16* gO = out + headoff + (size_t)qt * QT * GDIM;
  const int rbase = wave * 16 + (half ? 8 : 0);
#pragma unroll
  for (int i = 0; i < 8; ++i) {
    float inv = 1.0f / lrow[i];
#pragma unroll
    for (int g = 0; g < 8; ++g)
      gO[(size_t)(rbase + i) * GDIM + g * 16 + l16] = f2bf(accO[g][i] * inv);
  }
}

// =====================================================================
extern "C" void kernel_launch(void* const* d_in, const int* in_sizes, int n_in,
                              void* d_out, int out_size, void* d_ws, size_t ws_size,
                              hipStream_t stream) {
  const float* q  = (const float*)d_in[0];
  const float* k  = (const float*)d_in[1];
  const float* v  = (const float*)d_in[2];
  const float* wq = (const float*)d_in[3];
  const float* wk = (const float*)d_in[4];
  const float* wv = (const float*)d_in[5];
  const float* wo = (const float*)d_in[6];

  const size_t MS = (size_t)GM * GDIM;   // 8 Mi elems
  u16* xq   = (u16*)d_ws;
  u16* xk   = xq + MS;
  u16* xv   = xk + MS;
  u16* attn = xv + MS;

  dim3 gg(GN / BN, GM / BM);             // (16, 32)
  gemm_wmma<float, u16><<<gg, 256, 0, stream>>>(q, wq, xq);
  gemm_wmma<float, u16><<<gg, 256, 0, stream>>>(k, wk, xk);
  gemm_wmma<float, u16><<<gg, 256, 0, stream>>>(v, wv, xv);

  flash_attn<<<dim3(GS / QT, GB * GH), 128, 0, stream>>>(xq, xk, xv, attn);

  gemm_wmma<u16, float><<<gg, 256, 0, stream>>>(attn, wo, (float*)d_out);
}
